// Enhanced_GCN_with_Attention_11768210391289
// MI455X (gfx1250) — matrix operand order repaired and checked by reading, not compiled, not run on hardware
//
#include <hip/hip_runtime.h>
#include <hip/hip_bf16.h>

typedef __attribute__((ext_vector_type(16))) _Float16 v16h;
typedef __attribute__((ext_vector_type(8)))  _Float16 v8h;
typedef __attribute__((ext_vector_type(8)))  float    v8f;

#define GHEADS 4
#define GHID   128

enum { F_BIAS = 1, F_ACCUM = 2, F_RELU = 4 };

// ---------------------------------------------------------------------------
// WMMA GEMM:  C[MxN] = A[MxK] (f16 row-major) * B (staged as Bt[NxK] f16, i.e.
// column-major B), f32 accumulate via v_wmma_f32_16x16x32_f16.
// One wave computes one 16x16 tile; 4 waves per block cover 4 N-tiles.
// A-fragment per ISA: lane L (m=L%16, g=L/16) holds k = {kk+8g..+7, kk+8g+16..+23}
// B-fragment per ISA: lane L (n=L%16, g=L/16) holds k = {kk+8g..+7, kk+8g+16..+23}
// ---------------------------------------------------------------------------
__global__ void wmma_gemm_f16(const _Float16* __restrict__ A,
                              const _Float16* __restrict__ Bt,
                              const float* __restrict__ bias,
                              float* __restrict__ C,
                              int M, int N, int K, int flags)
{
  (void)M;
  const int lane = threadIdx.x;                    // 0..31
  const int tn = (blockIdx.x * blockDim.y + threadIdx.y) * 16;
  const int tm = blockIdx.y * 16;
  if (tn >= N) return;                             // wave-uniform
  const int r = lane & 15;
  const int g = lane >> 4;
  const _Float16* ap = A  + (size_t)(tm + r) * K + 8 * g;
  const _Float16* bp = Bt + (size_t)(tn + r) * K + 8 * g;

  v8f acc = {};
  for (int k = 0; k < K; k += 32) {
    v8h alo = *(const v8h*)(ap + k);        // k+8g .. k+8g+7
    v8h ahi = *(const v8h*)(ap + k + 16);   // k+8g+16 .. +23
    v8h blo = *(const v8h*)(bp + k);        // k+8g .. k+8g+7
    v8h bhi = *(const v8h*)(bp + k + 16);   // k+8g+16 .. +23
    v16h a = __builtin_shufflevector(alo, ahi, 0,1,2,3,4,5,6,7,8,9,10,11,12,13,14,15);
    v16h b = __builtin_shufflevector(blo, bhi, 0,1,2,3,4,5,6,7,8,9,10,11,12,13,14,15);
    acc = __builtin_amdgcn_wmma_f32_16x16x32_f16(false, a, false, b,
                                                 (short)0, acc, false, false);
  }

  const int col = tn + r;
  const float bv = (flags & F_BIAS) ? bias[col] : 0.0f;
#pragma unroll
  for (int i = 0; i < 8; ++i) {
    const int row = tm + i + 8 * g;                // C/D layout: VGPR i -> m=i+8g
    const size_t idx = (size_t)row * N + col;
    float v = acc[i] + bv;
    if (flags & F_ACCUM) v += C[idx];
    if (flags & F_RELU)  v = fmaxf(v, 0.0f);
    C[idx] = v;
  }
}

// ---------------------------------------------------------------------------
// Elementwise / staging kernels
// ---------------------------------------------------------------------------
__global__ void cvt_f16_kernel(const float* __restrict__ in, _Float16* __restrict__ out,
                               int n, int relu)
{
  int i = blockIdx.x * blockDim.x + threadIdx.x;
  if (i < n) {
    float v = in[i];
    if (relu) v = fmaxf(v, 0.0f);
    out[i] = (_Float16)v;
  }
}

// wt[n*K + k] = w[(rowoff + k)*N + n]   (stage weights transposed, f16)
__global__ void transpose_cvt_kernel(const float* __restrict__ w, _Float16* __restrict__ wt,
                                     int K, int N, int rowoff)
{
  int t = blockIdx.x * blockDim.x + threadIdx.x;
  if (t < K * N) {
    int k = t / N, n = t - k * N;
    wt[(size_t)n * K + k] = (_Float16)w[(size_t)(rowoff + k) * N + n];
  }
}

// ---------------------------------------------------------------------------
// Degree / symmetric norm (self-loops folded in as init=1)
// ---------------------------------------------------------------------------
__global__ void deg_init_kernel(float* deg, int n)
{ int i = blockIdx.x * blockDim.x + threadIdx.x; if (i < n) deg[i] = 1.0f; }

__global__ void deg_edges_kernel(const int* __restrict__ ei, float* deg, int E)
{ int e = blockIdx.x * blockDim.x + threadIdx.x; if (e < E) atomicAdd(&deg[ei[E + e]], 1.0f); }

__global__ void dinv_kernel(float* deg, int n)
{ int i = blockIdx.x * blockDim.x + threadIdx.x; if (i < n) deg[i] = rsqrtf(deg[i]); }

// out[i] = h[i]*dinv[i]^2 + bias   (self-loop message + bias)
__global__ void gcn_self_init_kernel(const float* __restrict__ h, const float* __restrict__ dinv,
                                     const float* __restrict__ bias, float* __restrict__ out, int C)
{
  int row = blockIdx.x, c = threadIdx.x;
  float di = dinv[row];
  out[(size_t)row * C + c] = h[(size_t)row * C + c] * di * di + bias[c];
}

__global__ void gcn_scatter_kernel(const int* __restrict__ ei, const float* __restrict__ h,
                                   const float* __restrict__ dinv, float* __restrict__ out,
                                   int E, int C)
{
  int e = blockIdx.x, c = threadIdx.x;
  int s = ei[e], d = ei[E + e];
  float w = dinv[s] * dinv[d];
  atomicAdd(&out[(size_t)d * C + c], h[(size_t)s * C + c] * w);
}

// ---------------------------------------------------------------------------
// GAT: attention logits, edge-softmax (max, sum, weighted scatter)
// ---------------------------------------------------------------------------
__device__ __forceinline__ float leaky02(float x) { return x > 0.0f ? x : 0.2f * x; }

__device__ __forceinline__ void atomicMaxF32(float* addr, float v)
{
  // ordered-int trick: works for any sign mix, init value is always a real float
  if (v >= 0.0f) atomicMax((int*)addr, __float_as_int(v));
  else           atomicMin((unsigned int*)addr, __float_as_uint(v));
}

// one wave per (node, head): a_src/a_dst dot products over 128 channels
__global__ void gat_att_kernel(const float* __restrict__ h2, const float* __restrict__ att_s,
                               const float* __restrict__ att_d, float* __restrict__ asrc,
                               float* __restrict__ adst)
{
  int nI = blockIdx.x;
  int h = threadIdx.y;
  int lane = threadIdx.x;
  const float* x = h2 + (size_t)nI * (GHEADS * GHID) + h * GHID;
  float s1 = 0.f, s2 = 0.f;
  for (int j = lane; j < GHID; j += 32) {
    float v = x[j];
    s1 += v * att_s[h * GHID + j];
    s2 += v * att_d[h * GHID + j];
  }
  for (int off = 16; off; off >>= 1) {
    s1 += __shfl_down(s1, off, 32);
    s2 += __shfl_down(s2, off, 32);
  }
  if (lane == 0) { asrc[nI * GHEADS + h] = s1; adst[nI * GHEADS + h] = s2; }
}

__global__ void gat_m_init_kernel(const float* asrc, const float* adst, float* m, int n4)
{
  int t = blockIdx.x * blockDim.x + threadIdx.x;
  if (t < n4) m[t] = leaky02(asrc[t] + adst[t]);          // self-loop logit
}

__global__ void gat_edge_max_kernel(const int* __restrict__ ei, const float* asrc,
                                    const float* adst, float* m, int E)
{
  int t = blockIdx.x * blockDim.x + threadIdx.x;
  if (t >= E * GHEADS) return;
  int e = t >> 2, h = t & 3;
  int s = ei[e], d = ei[E + e];
  atomicMaxF32(&m[d * GHEADS + h], leaky02(asrc[s * GHEADS + h] + adst[d * GHEADS + h]));
}

__global__ void gat_z_init_kernel(const float* asrc, const float* adst, const float* m,
                                  float* z, int n4)
{
  int t = blockIdx.x * blockDim.x + threadIdx.x;
  if (t < n4) z[t] = __expf(leaky02(asrc[t] + adst[t]) - m[t]);
}

__global__ void gat_edge_z_kernel(const int* __restrict__ ei, const float* asrc,
                                  const float* adst, const float* m, float* z, int E)
{
  int t = blockIdx.x * blockDim.x + threadIdx.x;
  if (t >= E * GHEADS) return;
  int e = t >> 2, h = t & 3;
  int s = ei[e], d = ei[E + e];
  float v = leaky02(asrc[s * GHEADS + h] + adst[d * GHEADS + h]);
  atomicAdd(&z[d * GHEADS + h], __expf(v - m[d * GHEADS + h]));
}

// out[n][c] = h2[n][c]*alpha_self + gat_b[c]   (512 threads per node)
__global__ void gat_self_init_kernel(const float* __restrict__ h2, const float* asrc,
                                     const float* adst, const float* m, const float* z,
                                     const float* __restrict__ gb, float* __restrict__ out)
{
  int nI = blockIdx.x, c = threadIdx.x;
  int h = c >> 7;
  float e = leaky02(asrc[nI * GHEADS + h] + adst[nI * GHEADS + h]);
  float alpha = __expf(e - m[nI * GHEADS + h]) / z[nI * GHEADS + h];
  size_t idx = (size_t)nI * 512 + c;
  out[idx] = h2[idx] * alpha + gb[c];
}

__global__ void gat_scatter_kernel(const int* __restrict__ ei, const float* __restrict__ h2,
                                   const float* asrc, const float* adst, const float* m,
                                   const float* z, float* __restrict__ out, int E)
{
  int e = blockIdx.x, c = threadIdx.x;
  int h = c >> 7;
  int s = ei[e], d = ei[E + e];
  float ev = leaky02(asrc[s * GHEADS + h] + adst[d * GHEADS + h]);
  float alpha = __expf(ev - m[d * GHEADS + h]) / z[d * GHEADS + h];
  atomicAdd(&out[(size_t)d * 512 + c], h2[(size_t)s * 512 + c] * alpha);
}

// ---------------------------------------------------------------------------
// LayerNorm (optionally fused ReLU), writes f16 staging for the next GEMM
// ---------------------------------------------------------------------------
__global__ void layernorm_f16_kernel(const float* __restrict__ in, const float* __restrict__ g,
                                     const float* __restrict__ b, _Float16* __restrict__ out,
                                     int C, int relu)
{
  __shared__ float s1[256], s2[256];
  int row = blockIdx.x, t = threadIdx.x;
  const float* x = in + (size_t)row * C;
  float a = 0.f, aa = 0.f;
  for (int j = t; j < C; j += 256) { float v = x[j]; a += v; aa += v * v; }
  s1[t] = a; s2[t] = aa; __syncthreads();
  for (int st = 128; st; st >>= 1) {
    if (t < st) { s1[t] += s1[t + st]; s2[t] += s2[t + st]; }
    __syncthreads();
  }
  float mu  = s1[0] / C;
  float var = s2[0] / C - mu * mu;
  float rs  = rsqrtf(var + 1e-5f);
  for (int j = t; j < C; j += 256) {
    float v = (x[j] - mu) * rs * g[j] + b[j];
    if (relu) v = fmaxf(v, 0.0f);
    out[(size_t)row * C + j] = (_Float16)v;
  }
}

// ---------------------------------------------------------------------------
// Final 128 -> 2 head (block per row, shared-mem reduce)
// ---------------------------------------------------------------------------
__global__ void mlp2_kernel(const float* __restrict__ hid, const float* __restrict__ w,
                            const float* __restrict__ b, float* __restrict__ out, int n)
{
  (void)n;
  __shared__ float r0[128], r1[128];
  int row = blockIdx.x, t = threadIdx.x;
  float v = hid[(size_t)row * 128 + t];
  r0[t] = v * w[t * 2 + 0];
  r1[t] = v * w[t * 2 + 1];
  __syncthreads();
  for (int st = 64; st; st >>= 1) {
    if (t < st) { r0[t] += r0[t + st]; r1[t] += r1[t + st]; }
    __syncthreads();
  }
  if (t == 0) {
    out[(size_t)row * 2 + 0] = r0[0] + b[0];
    out[(size_t)row * 2 + 1] = r1[0] + b[1];
  }
}

// ---------------------------------------------------------------------------
extern "C" void kernel_launch(void* const* d_in, const int* in_sizes, int n_in,
                              void* d_out, int out_size, void* d_ws, size_t ws_size,
                              hipStream_t stream)
{
  (void)n_in; (void)out_size; (void)ws_size;
  const float* x       = (const float*)d_in[0];
  const int*   ei      = (const int*)  d_in[1];
  const float* gcn1_w  = (const float*)d_in[2];
  const float* gcn1_b  = (const float*)d_in[3];
  const float* gat_w   = (const float*)d_in[4];
  const float* att_src = (const float*)d_in[5];
  const float* att_dst = (const float*)d_in[6];
  const float* gat_b   = (const float*)d_in[7];
  const float* n1_g    = (const float*)d_in[8];
  const float* n1_b    = (const float*)d_in[9];
  const float* gcn2_w  = (const float*)d_in[10];
  const float* gcn2_b  = (const float*)d_in[11];
  const float* n2_g    = (const float*)d_in[12];
  const float* n2_b    = (const float*)d_in[13];
  const float* mlp_w1  = (const float*)d_in[14];
  const float* mlp_b1  = (const float*)d_in[15];
  const float* mlp_w2  = (const float*)d_in[16];
  const float* mlp_b2  = (const float*)d_in[17];

  const int Nn = in_sizes[0] / 128;   // 50000 (multiple of 16)
  const int E  = in_sizes[1] / 2;     // 800000
  const int IN_C = 128, HH = 512, G2 = 256, MH = 128;

  char* ws = (char*)d_ws;
  size_t off = 0;
  auto alloc = [&](size_t elems, size_t esz) -> void* {
    void* p = ws + off;
    off += (elems * esz + 255) & ~(size_t)255;
    return p;
  };
  _Float16* x_h   = (_Float16*)alloc((size_t)Nn * IN_C, 2);
  _Float16* h1_h  = (_Float16*)alloc((size_t)Nn * GHID, 2);
  _Float16* ln1_h = (_Float16*)alloc((size_t)Nn * HH, 2);
  _Float16* ln2_h = (_Float16*)alloc((size_t)Nn * G2, 2);
  _Float16* w1t   = (_Float16*)alloc(128 * 128, 2);
  _Float16* wgt   = (_Float16*)alloc(512 * 128, 2);
  _Float16* w2t   = (_Float16*)alloc(256 * 512, 2);
  _Float16* wm1at = (_Float16*)alloc(128 * 256, 2);
  _Float16* wm1bt = (_Float16*)alloc(128 * 128, 2);
  float* hlin1  = (float*)alloc((size_t)Nn * GHID, 4);
  float* h1f    = (float*)alloc((size_t)Nn * GHID, 4);
  float* h2f    = (float*)alloc((size_t)Nn * HH, 4);
  float* gatout = (float*)alloc((size_t)Nn * HH, 4);
  float* g2lin  = (float*)alloc((size_t)Nn * G2, 4);
  float* g2out  = (float*)alloc((size_t)Nn * G2, 4);
  float* mlp1   = (float*)alloc((size_t)Nn * MH, 4);
  float* asrc   = (float*)alloc((size_t)Nn * 4, 4);
  float* adst   = (float*)alloc((size_t)Nn * 4, 4);
  float* mbuf   = (float*)alloc((size_t)Nn * 4, 4);
  float* zbuf   = (float*)alloc((size_t)Nn * 4, 4);
  float* dinv   = (float*)alloc((size_t)Nn, 4);

  auto cdiv = [](int a, int b) { return (a + b - 1) / b; };
  const dim3 gblk(32, 4);                 // 4 waves per block, 1 N-tile each
  const int tilesM = Nn / 16;

  // stage weights (transposed, f16) -- all fit in L2 with ease
  transpose_cvt_kernel<<<cdiv(128 * 128, 256), 256, 0, stream>>>(gcn1_w, w1t, 128, 128, 0);
  transpose_cvt_kernel<<<cdiv(128 * 512, 256), 256, 0, stream>>>(gat_w,  wgt, 128, 512, 0);
  transpose_cvt_kernel<<<cdiv(512 * 256, 256), 256, 0, stream>>>(gcn2_w, w2t, 512, 256, 0);
  transpose_cvt_kernel<<<cdiv(256 * 128, 256), 256, 0, stream>>>(mlp_w1, wm1at, 256, 128, 0);
  transpose_cvt_kernel<<<cdiv(128 * 128, 256), 256, 0, stream>>>(mlp_w1, wm1bt, 128, 128, 256);

  // symmetric degree norm (self-loops => init 1)
  deg_init_kernel <<<cdiv(Nn, 256), 256, 0, stream>>>(dinv, Nn);
  deg_edges_kernel<<<cdiv(E, 256), 256, 0, stream>>>(ei, dinv, E);
  dinv_kernel     <<<cdiv(Nn, 256), 256, 0, stream>>>(dinv, Nn);

  // stage x as f16 (used by GCN1 GEMM and MLP1 part b)
  cvt_f16_kernel<<<cdiv(Nn * IN_C, 256), 256, 0, stream>>>(x, x_h, Nn * IN_C, 0);

  // --- GCN1: h = scatter(xW) + b, ReLU ---
  wmma_gemm_f16<<<dim3(8 / 4, tilesM), gblk, 0, stream>>>(x_h, w1t, nullptr, hlin1, Nn, 128, 128, 0);
  gcn_self_init_kernel<<<Nn, 128, 0, stream>>>(hlin1, dinv, gcn1_b, h1f, 128);
  gcn_scatter_kernel  <<<E, 128, 0, stream>>>(ei, hlin1, dinv, h1f, E, 128);
  cvt_f16_kernel<<<cdiv(Nn * GHID, 256), 256, 0, stream>>>(h1f, h1_h, Nn * GHID, 1); // fused ReLU

  // --- GAT: h2 = hW (N x 512), additive attention + edge softmax ---
  wmma_gemm_f16<<<dim3(32 / 4, tilesM), gblk, 0, stream>>>(h1_h, wgt, nullptr, h2f, Nn, 512, 128, 0);
  gat_att_kernel<<<Nn, dim3(32, GHEADS), 0, stream>>>(h2f, att_src, att_dst, asrc, adst);
  gat_m_init_kernel  <<<cdiv(Nn * 4, 256), 256, 0, stream>>>(asrc, adst, mbuf, Nn * 4);
  gat_edge_max_kernel<<<cdiv(E * 4, 256), 256, 0, stream>>>(ei, asrc, adst, mbuf, E);
  gat_z_init_kernel  <<<cdiv(Nn * 4, 256), 256, 0, stream>>>(asrc, adst, mbuf, zbuf, Nn * 4);
  gat_edge_z_kernel  <<<cdiv(E * 4, 256), 256, 0, stream>>>(ei, asrc, adst, mbuf, zbuf, E);
  gat_self_init_kernel<<<Nn, 512, 0, stream>>>(h2f, asrc, adst, mbuf, zbuf, gat_b, gatout);
  gat_scatter_kernel  <<<E, 512, 0, stream>>>(ei, h2f, asrc, adst, mbuf, zbuf, gatout, E);

  // --- LN1 -> f16 staging ---
  layernorm_f16_kernel<<<Nn, 256, 0, stream>>>(gatout, n1_g, n1_b, ln1_h, 512, 0);

  // --- GCN2: scatter(ln1 @ W2) + b ---
  wmma_gemm_f16<<<dim3(16 / 4, tilesM), gblk, 0, stream>>>(ln1_h, w2t, nullptr, g2lin, Nn, 256, 512, 0);
  gcn_self_init_kernel<<<Nn, 256, 0, stream>>>(g2lin, dinv, gcn2_b, g2out, 256);
  gcn_scatter_kernel  <<<E, 256, 0, stream>>>(ei, g2lin, dinv, g2out, E, 256);

  // --- LN2 + ReLU -> f16 staging ---
  layernorm_f16_kernel<<<Nn, 256, 0, stream>>>(g2out, n2_g, n2_b, ln2_h, 256, 1);

  // --- MLP1 on implicit concat [h256 | x128]: two GEMMs, second accumulates + ReLU ---
  wmma_gemm_f16<<<dim3(8 / 4, tilesM), gblk, 0, stream>>>(ln2_h, wm1at, mlp_b1, mlp1, Nn, 128, 256, F_BIAS);
  wmma_gemm_f16<<<dim3(8 / 4, tilesM), gblk, 0, stream>>>(x_h,   wm1bt, nullptr, mlp1, Nn, 128, 128, F_ACCUM | F_RELU);

  // --- MLP2 head ---
  mlp2_kernel<<<Nn, 128, 0, stream>>>(mlp1, mlp_w2, mlp_b2, (float*)d_out, Nn);
}
